// EncoderRNN_70849780515145
// MI455X (gfx1250) — matrix operand order repaired and checked by reading, not compiled, not run on hardware
//
#include <hip/hip_runtime.h>
#include <hip/hip_bf16.h>

// ---------------------------------------------------------------------------
// LSTM encoder for MI455X (gfx1250).  T=512, B=64, I=300, H=1000, G=4000.
//
// Persistent kernel, 125 WGs x 256 threads (8 waves, wave32). Each WG owns
// SLICE=8 hidden columns (32 gate columns). Per step each wave computes one
// 16x16 f32 tile of  gates = x_t @ W_ih^T + h @ W_hh^T  via
// v_wmma_f32_16x16x32_f16 (f16 operands, f32 accum, dual accumulators).
//   - W_hh slice staged in LDS once (66 KB, padded stride -> bank-conflict-free)
//   - h staged in LDS per step via global_load_async_to_lds_b128 (ASYNCcnt)
//   - cell state c in LDS; timestep sync = agent-scope release/acquire counter
//   - h double-buffered f16 [2][64][1024] in workspace (K padded to 1024)
// ---------------------------------------------------------------------------

typedef __attribute__((ext_vector_type(16))) _Float16 v16h;
typedef __attribute__((ext_vector_type(8)))  _Float16 v8h;
typedef __attribute__((ext_vector_type(8)))  float    v8f;

#define T_STEPS 512
#define BATCH   64
#define IN_DIM  300
#define IN_PAD  320
#define HID     1000
#define HID_PAD 1024
#define GATES   4000
#define SLICE   8
#define NWG     (HID / SLICE)   // 125 workgroups
#define LDH     1032            // LDS row stride (halves): 2064B -> banks 4n mod 64

// dynamic LDS carve
#define SM_HL_BYTES   (BATCH * LDH * 2)          // 132096
#define SM_WL_BYTES   (32 * LDH * 2)             // 66048
#define SM_GATE_BYTES (BATCH * 32 * 4)           // 8192
#define SM_C_BYTES    (BATCH * SLICE * 4)        // 2048
#define SM_BIAS_BYTES (32 * 4)                   // 128
#define SM_TOTAL (SM_HL_BYTES + SM_WL_BYTES + SM_GATE_BYTES + SM_C_BYTES + SM_BIAS_BYTES)

// ---- CDNA5 async global->LDS copy (ASYNCcnt-tracked), ISA 15.18.3 ----------
__device__ __forceinline__ void async_ld_b128(unsigned lds_off, const void* gptr) {
  asm volatile("global_load_async_to_lds_b128 %0, %1, off"
               :: "v"(lds_off), "v"(gptr) : "memory");
}
__device__ __forceinline__ void wait_async0() {
  asm volatile("s_wait_asynccnt 0x0" ::: "memory");
}

// ---------------- fragment loads (wave32 WMMA layouts, ISA 7.12.2) ----------

// A 16x32 f16: lanes 0-15: M=lane, K=k0+[0..7] (v0-3), k0+[16..23] (v4-7);
// lanes 16-31: M=lane-16, K=k0+[8..15], k0+[24..31].
__device__ __forceinline__ v16h load_a_frag(const _Float16* base, int ld,
                                            int m0, int k0, int lane) {
  const int hi = lane >> 4;
  const int m  = m0 + (lane & 15);
  const _Float16* p = base + (size_t)m * ld + k0 + hi * 8;
  v8h lo = *(const v8h*)p;
  v8h up = *(const v8h*)(p + 16);
  v16h r;
#pragma unroll
  for (int i = 0; i < 8; ++i) { r[i] = lo[i]; r[i + 8] = up[i]; }
  return r;
}

// B 32x16 f16: lanes 0-15: col N=lane, K=k0+[0..7], k0+[16..23]; lanes 16-31: col N=lane-16,
// K=k0+[8..15], k0+[24..31].  rowp = start of the K-contiguous row for this lane's column.
__device__ __forceinline__ v16h load_b_frag(const _Float16* rowp, int k0, int lane) {
  const int hi = lane >> 4;
  const _Float16* p = rowp + k0 + hi * 8;
  v8h lo = *(const v8h*)p;            // 16B aligned (LDS stride 2064B)
  v8h up = *(const v8h*)(p + 16);
  v16h r;
#pragma unroll
  for (int i = 0; i < 8; ++i) { r[i] = lo[i]; r[i + 8] = up[i]; }
  return r;
}

__device__ __forceinline__ float sigmoidf_(float x) {
  return 1.0f / (1.0f + expf(-x));
}

// ---------------- prep kernels ---------------------------------------------

__global__ void pad_convert_f16(const float* __restrict__ src,
                                _Float16* __restrict__ dst,
                                int rows, int cs, int cd) {
  long long n = (long long)rows * cd;
  for (long long i = (long long)blockIdx.x * blockDim.x + threadIdx.x; i < n;
       i += (long long)gridDim.x * blockDim.x) {
    int r = (int)(i / cd);
    int k = (int)(i % cd);
    dst[i] = (k < cs) ? (_Float16)src[(long long)r * cs + k] : (_Float16)0.0f;
  }
}

__global__ void zero_f16(_Float16* __restrict__ p, int n) {
  for (int i = blockIdx.x * blockDim.x + threadIdx.x; i < n;
       i += gridDim.x * blockDim.x)
    p[i] = (_Float16)0.0f;
}

__global__ void bias_bar_init(const float* __restrict__ bih,
                              const float* __restrict__ bhh,
                              float* __restrict__ bias,
                              unsigned* __restrict__ bar, int n) {
  int i = blockIdx.x * blockDim.x + threadIdx.x;
  if (i < n) bias[i] = bih[i] + bhh[i];
  if (i == 0) *bar = 0u;
}

// ---------------- persistent LSTM kernel -----------------------------------

__global__ __launch_bounds__(256) void lstm_persistent(
    const _Float16* __restrict__ x16,    // [T][64][IN_PAD]
    const _Float16* __restrict__ Wih16,  // [GATES][IN_PAD]
    const _Float16* __restrict__ Whh16,  // [GATES][HID_PAD]
    const float*    __restrict__ bias,   // [GATES] = b_ih + b_hh
    const float*    __restrict__ c0,     // [64][HID]
    _Float16*       __restrict__ hbuf,   // [2][64][HID_PAD] double buffer
    float*          __restrict__ out,    // [T][64][HID]
    unsigned*       __restrict__ bar) {
  extern __shared__ char smem[];
  _Float16* hL    = (_Float16*)smem;                              // [64][LDH]
  _Float16* wL    = (_Float16*)(smem + SM_HL_BYTES);              // [32][LDH]
  float*    gatesL= (float*)(smem + SM_HL_BYTES + SM_WL_BYTES);   // [64][32]
  float*    cL    = gatesL + BATCH * 32;                          // [64][8]
  float*    biasL = cL + BATCH * SLICE;                           // [32]

  const int tid  = threadIdx.x;
  const int wave = tid >> 5;
  const int lane = tid & 31;
  const int m0   = (wave & 3) * 16;      // batch-tile origin
  const int n0   = (wave >> 2) * 16;     // gate-column-tile origin
  const int j0   = blockIdx.x * SLICE;   // hidden-column slice origin

  // Per-lane gate column for this wave's tile (local index 0..31)
  const int nloc = n0 + (lane & 15);
  const int browI = (nloc >> 3) * HID + j0 + (nloc & 7);   // row of W_ih
  const _Float16* Bih = Wih16 + (size_t)browI * IN_PAD;
  const _Float16* BhhL = wL + nloc * LDH;                  // W_hh row in LDS

  if (tid < 32) biasL[tid] = bias[(tid >> 3) * HID + j0 + (tid & 7)];
  for (int i = tid; i < BATCH * SLICE; i += 256) {
    int b = i >> 3, s = i & 7;
    cL[i] = c0[b * HID + j0 + s];
  }
  // Stage this WG's 32 W_hh rows into LDS once (32 x 1024 halves).
  for (int i = tid; i < 32 * 128; i += 256) {
    int n = i >> 7;                   // local gate column 0..31
    int kc = (i & 127) * 8;           // 8-half chunk
    int grow = (n >> 3) * HID + j0 + (n & 7);
    *(v8h*)(wL + n * LDH + kc) = *(const v8h*)(Whh16 + (size_t)grow * HID_PAD + kc);
  }
  __syncthreads();

  for (int t = 0; t < T_STEPS; ++t) {
    const _Float16* Ax = x16 + (size_t)t * BATCH * IN_PAD;
    const _Float16* Ah = hbuf + (size_t)(t & 1) * BATCH * HID_PAD;
    _Float16* Hw = hbuf + (size_t)((t & 1) ^ 1) * BATCH * HID_PAD;

    // ---- stage h (64x1024 f16) into LDS via async global->LDS b128 ----
#pragma unroll 4
    for (int i = 0; i < 32; ++i) {
      int chunk = i * 256 + tid;      // 0..8191
      int row = chunk >> 7;
      int kc  = (chunk & 127) * 8;
      unsigned loff = (unsigned)(size_t)(hL + row * LDH + kc);
      async_ld_b128(loff, Ah + (size_t)row * HID_PAD + kc);
    }
    if (t + 1 < T_STEPS)  // hint next x tile toward L2/WGP$
      __builtin_prefetch(Ax + BATCH * IN_PAD + tid * 80, 0, 0);
    wait_async0();
    __syncthreads();

    // ---- WMMA: dual accumulators to expose matrix-pipe ILP ----
    v8f acc0 = {0.f,0.f,0.f,0.f,0.f,0.f,0.f,0.f};
    v8f acc1 = {0.f,0.f,0.f,0.f,0.f,0.f,0.f,0.f};

    // input projection: K = 320 (padded), 10 WMMA (A from global, B hoistable)
#pragma unroll
    for (int k0 = 0; k0 < IN_PAD; k0 += 64) {
      v16h a0 = load_a_frag(Ax, IN_PAD, m0, k0, lane);
      v16h b0 = load_b_frag(Bih, k0, lane);
      acc0 = __builtin_amdgcn_wmma_f32_16x16x32_f16(false, a0, false, b0,
                                                    (short)0, acc0, false, false);
      v16h a1 = load_a_frag(Ax, IN_PAD, m0, k0 + 32, lane);
      v16h b1 = load_b_frag(Bih, k0 + 32, lane);
      acc1 = __builtin_amdgcn_wmma_f32_16x16x32_f16(false, a1, false, b1,
                                                    (short)0, acc1, false, false);
    }
    // recurrent projection: K = 1024 (padded), 32 WMMA, A/B from LDS
#pragma unroll 4
    for (int k0 = 0; k0 < HID_PAD; k0 += 64) {
      v16h a0 = load_a_frag(hL, LDH, m0, k0, lane);
      v16h b0 = load_b_frag(BhhL, k0, lane);
      acc0 = __builtin_amdgcn_wmma_f32_16x16x32_f16(false, a0, false, b0,
                                                    (short)0, acc0, false, false);
      v16h a1 = load_a_frag(hL, LDH, m0, k0 + 32, lane);
      v16h b1 = load_b_frag(BhhL, k0 + 32, lane);
      acc1 = __builtin_amdgcn_wmma_f32_16x16x32_f16(false, a1, false, b1,
                                                    (short)0, acc1, false, false);
    }

    // C layout: VGPR r -> lanes 0-15: (M=m0+r, N=n0+lane); lanes 16-31: M+8.
    {
      const int row = m0 + ((lane >> 4) ? 8 : 0);
      const int col = n0 + (lane & 15);
#pragma unroll
      for (int r = 0; r < 8; ++r)
        gatesL[(row + r) * 32 + col] = acc0[r] + acc1[r];
    }
    __syncthreads();

    // ---- elementwise LSTM cell: 64*8 = 512 cells, 2 per thread ----
#pragma unroll
    for (int c2 = 0; c2 < 2; ++c2) {
      const int cell = tid + c2 * 256;
      const int b = cell >> 3, s = cell & 7;
      float iv = sigmoidf_(gatesL[b * 32 + s]      + biasL[s]);
      float fv = sigmoidf_(gatesL[b * 32 + 8 + s]  + biasL[8 + s]);
      float gv = tanhf    (gatesL[b * 32 + 16 + s] + biasL[16 + s]);
      float ov = sigmoidf_(gatesL[b * 32 + 24 + s] + biasL[24 + s]);
      float cn = fv * cL[cell] + iv * gv;
      cL[cell] = cn;
      float hn = ov * tanhf(cn);
      out[(size_t)t * (BATCH * HID) + b * HID + j0 + s] = hn;
      Hw[b * HID_PAD + j0 + s] = (_Float16)hn;
    }

    // ---- device-wide step barrier: release stores, arrive, spin, acquire ----
    __threadfence();
    __syncthreads();
    if (tid == 0) {
      __hip_atomic_fetch_add(bar, 1u, __ATOMIC_RELEASE,
                             __HIP_MEMORY_SCOPE_AGENT);
      const unsigned target = (unsigned)(t + 1) * NWG;
      while (__hip_atomic_load(bar, __ATOMIC_ACQUIRE,
                               __HIP_MEMORY_SCOPE_AGENT) < target)
        __builtin_amdgcn_s_sleep(2);
    }
    __syncthreads();
    __builtin_amdgcn_fence(__ATOMIC_ACQUIRE, "agent");  // inv stale WGP$ lines
  }
}

// ---------------- host launcher --------------------------------------------

extern "C" void kernel_launch(void* const* d_in, const int* in_sizes, int n_in,
                              void* d_out, int out_size, void* d_ws,
                              size_t ws_size, hipStream_t stream) {
  const float* inputs = (const float*)d_in[0];  // [512,64,300]
  const float* h0     = (const float*)d_in[1];  // [64,1000]
  const float* c0     = (const float*)d_in[2];  // [64,1000]
  const float* W_ih   = (const float*)d_in[3];  // [4000,300]
  const float* W_hh   = (const float*)d_in[4];  // [4000,1000]
  const float* b_ih   = (const float*)d_in[5];  // [4000]
  const float* b_hh   = (const float*)d_in[6];  // [4000]
  float* out = (float*)d_out;                   // [512,64,1000]

  // workspace layout (256B aligned), total ~32 MB
  char* ws = (char*)d_ws;
  const size_t OFF_BAR = 0;
  const size_t OFF_H   = 256;                                   // 2*64*1024*2
  const size_t OFF_WHH = OFF_H + 2ull * BATCH * HID_PAD * 2;    // 4000*1024*2
  const size_t OFF_WIH = OFF_WHH + (size_t)GATES * HID_PAD * 2; // 4000*320*2
  const size_t OFF_X   = OFF_WIH + (size_t)GATES * IN_PAD * 2;  // T*64*320*2
  const size_t OFF_B   = OFF_X + (size_t)T_STEPS * BATCH * IN_PAD * 2;

  unsigned*  bar   = (unsigned*)(ws + OFF_BAR);
  _Float16*  hbuf  = (_Float16*)(ws + OFF_H);
  _Float16*  Whh16 = (_Float16*)(ws + OFF_WHH);
  _Float16*  Wih16 = (_Float16*)(ws + OFF_WIH);
  _Float16*  x16   = (_Float16*)(ws + OFF_X);
  float*     bias  = (float*)(ws + OFF_B);

  auto blocks = [](long long n) {
    long long b = (n + 255) / 256;
    return (int)(b > 8192 ? 8192 : b);
  };

  // prep: f16 conversions with K padding, h0/c0/bias/barrier init
  {
    int n = 2 * BATCH * HID_PAD;
    zero_f16<<<blocks(n), 256, 0, stream>>>(hbuf, n);
  }
  pad_convert_f16<<<blocks((long long)BATCH * HID_PAD), 256, 0, stream>>>(
      h0, hbuf, BATCH, HID, HID_PAD);  // buffer 0 <- h0 (padding zeroed)
  pad_convert_f16<<<blocks((long long)GATES * HID_PAD), 256, 0, stream>>>(
      W_hh, Whh16, GATES, HID, HID_PAD);
  pad_convert_f16<<<blocks((long long)GATES * IN_PAD), 256, 0, stream>>>(
      W_ih, Wih16, GATES, IN_DIM, IN_PAD);
  pad_convert_f16<<<blocks((long long)T_STEPS * BATCH * IN_PAD), 256, 0,
                    stream>>>(inputs, x16, T_STEPS * BATCH, IN_DIM, IN_PAD);
  bias_bar_init<<<blocks(GATES), 256, 0, stream>>>(b_ih, b_hh, bias, bar,
                                                   GATES);

  // persistent recurrence: 125 co-resident WGs (204KB dynamic LDS each -> one
  // per WGP), in-kernel timestep barrier
  lstm_persistent<<<NWG, 256, SM_TOTAL, stream>>>(x16, Wih16, Whh16, bias, c0,
                                                  hbuf, out, bar);
  (void)in_sizes; (void)n_in; (void)out_size; (void)ws_size;
}
